// TriangleMultiplicativeUpdate_6305011990747
// MI455X (gfx1250) — matrix operand order repaired and checked by reading, compiled, not run on hardware
//
#include <hip/hip_runtime.h>

// ---------------------------------------------------------------------------
// Types / helpers
// ---------------------------------------------------------------------------
typedef __bf16 bf16_t;
typedef bf16_t bf16x16 __attribute__((ext_vector_type(16)));
typedef float  f32x8   __attribute__((ext_vector_type(8)));

#define DEVINL __device__ __forceinline__

struct alignas(16) U4 { unsigned a, b, c, d; };

union Frag { bf16x16 v; U4 q[2]; };

DEVINL bf16_t f2bf(float f) {
  union { float f; unsigned u; } v; v.f = f;
  unsigned r = v.u + 0x7fffu + ((v.u >> 16) & 1u);   // round-to-nearest-even
  unsigned short h = (unsigned short)(r >> 16);
  bf16_t out;
  __builtin_memcpy(&out, &h, 2);
  return out;
}

DEVINL float bf2f(bf16_t h) {
  unsigned short u;
  __builtin_memcpy(&u, &h, 2);
  union { unsigned u; float f; } v; v.u = ((unsigned)u) << 16;
  return v.f;
}

DEVINL float sigmoidf_(float x) { return 1.0f / (1.0f + __expf(-x)); }

DEVINL f32x8 zero8() {
  f32x8 r;
#pragma unroll
  for (int i = 0; i < 8; ++i) r[i] = 0.0f;
  return r;
}

// A-matrix 16x32 bf16 fragment (ISA 7.12.2): lane&15 = M row;
// lanes 0-15 hold K {0..7,16..23}, lanes 16-31 hold K {8..15,24..31}.
DEVINL bf16x16 load_frag_a(const bf16_t* base, int ld) {
  unsigned lane = threadIdx.x & 31u;
  const bf16_t* p = base + (lane & 15u) * ld + ((lane >> 4) ? 8u : 0u);
  Frag f;
  f.q[0] = *reinterpret_cast<const U4*>(p);
  f.q[1] = *reinterpret_cast<const U4*>(p + 16);
  return f.v;
}

// B-matrix 32x16 bf16 fragment: lane&15 = N column; lanes 0-15 hold K {0..7,16..23},
// lanes 16-31 hold K {8..15,24..31} (same order as A). LDS layout is [N][K].
DEVINL bf16x16 load_frag_b(const bf16_t* base, int ld) {
  unsigned lane = threadIdx.x & 31u;
  const bf16_t* p = base + (lane & 15u) * ld + ((lane >> 4) ? 8u : 0u);
  Frag f;
  f.q[0] = *reinterpret_cast<const U4*>(p);
  f.q[1] = *reinterpret_cast<const U4*>(p + 16);
  return f.v;
}

DEVINL f32x8 wmma_bf16(bf16x16 a, bf16x16 b, f32x8 c) {
  return __builtin_amdgcn_wmma_f32_16x16x32_bf16(
      /*neg_a=*/false, a, /*neg_b=*/false, b,
      /*c_mod=*/(short)0, c, /*reuse_a=*/false, /*reuse_b=*/false);
}

// Async direct global->LDS copy of 16B (+16B at offset:16), ASYNCcnt-tracked.
// GVS addressing: 64-bit SGPR base + per-lane 32-bit byte offset (ISA 10.2).
// The 24-bit inst offset is added on BOTH the LDS and global side (ISA 15.18.3).
DEVINL void async_copy_32B(unsigned lds_addr, unsigned gvoff_bytes,
                           unsigned long long sbase) {
  asm volatile("global_load_async_to_lds_b128 %0, %1, %2 offset:0"
               :: "v"(lds_addr), "v"(gvoff_bytes), "s"(sbase) : "memory");
  asm volatile("global_load_async_to_lds_b128 %0, %1, %2 offset:16"
               :: "v"(lds_addr), "v"(gvoff_bytes), "s"(sbase) : "memory");
}

DEVINL void wait_async0() {
  asm volatile("s_wait_asynccnt 0x0" ::: "memory");
}

// low 32 bits of a generic pointer to __shared__ == LDS byte offset
DEVINL unsigned lds_addr_of(const void* p) {
  return (unsigned)(unsigned long long)p;
}

// ---------------------------------------------------------------------------
// Constants
// ---------------------------------------------------------------------------
static constexpr int N      = 512;         // sequence
static constexpr int C      = 128;         // channels (C_Z == C_H == 128)
static constexpr int S      = N * N;       // 262144 spatial positions
static constexpr int MROWS  = 64;          // spatial rows per block
static constexpr float LNEPS = 1e-5f;

// ---------------------------------------------------------------------------
// Kernel 1: LayerNorm(z) + 5 projections (WMMA bf16)
//   a_ws, b_ws : bf16, channel-major [c][s]   (s = i*512+k)
//   g_ws       : bf16, [s][c]
// ---------------------------------------------------------------------------
__global__ __launch_bounds__(256)
void tmu_proj_kernel(const float* __restrict__ z, const float* __restrict__ mask,
                     const float* __restrict__ ln_w, const float* __restrict__ ln_b,
                     const float* __restrict__ w_ag, const float* __restrict__ b_ag,
                     const float* __restrict__ w_ap, const float* __restrict__ b_ap,
                     const float* __restrict__ w_bg, const float* __restrict__ b_bg,
                     const float* __restrict__ w_bp, const float* __restrict__ b_bp,
                     const float* __restrict__ w_g,  const float* __restrict__ b_g,
                     bf16_t* __restrict__ a_ws, bf16_t* __restrict__ b_ws,
                     bf16_t* __restrict__ g_ws) {
  __shared__ __align__(16) bf16_t zn[MROWS][C];   // 16 KB  (A operand, [M][K])
  __shared__ __align__(16) bf16_t wT[C][C];       // 32 KB  (B operand, [N][K])
  __shared__ __align__(16) float  tmp[MROWS][C];  // 32 KB  (transpose staging)

  const int sbase = blockIdx.x * MROWS;
  const int tid   = threadIdx.x;
  const int wave  = tid >> 5;
  const int lane  = tid & 31;

  // ---- LayerNorm of 64 input rows -> zn (bf16), one row per wave ----
#pragma unroll
  for (int it = 0; it < 8; ++it) {
    const int row = it * 8 + wave;
    const float* zr = z + (size_t)(sbase + row) * C;
    float4 v = *reinterpret_cast<const float4*>(zr + lane * 4);
    float s  = v.x + v.y + v.z + v.w;
    float sq = v.x * v.x + v.y * v.y + v.z * v.z + v.w * v.w;
#pragma unroll
    for (int off = 16; off > 0; off >>= 1) {
      s  += __shfl_xor(s,  off, 32);
      sq += __shfl_xor(sq, off, 32);
    }
    const float mu  = s * (1.0f / C);
    const float var = sq * (1.0f / C) - mu * mu;
    const float rs  = rsqrtf(var + LNEPS);
    float4 w4 = *reinterpret_cast<const float4*>(ln_w + lane * 4);
    float4 b4 = *reinterpret_cast<const float4*>(ln_b + lane * 4);
    zn[row][lane * 4 + 0] = f2bf((v.x - mu) * rs * w4.x + b4.x);
    zn[row][lane * 4 + 1] = f2bf((v.y - mu) * rs * w4.y + b4.y);
    zn[row][lane * 4 + 2] = f2bf((v.z - mu) * rs * w4.z + b4.z);
    zn[row][lane * 4 + 3] = f2bf((v.w - mu) * rs * w4.w + b4.w);
  }
  __syncthreads();

  // stage a 128x128 weight, transposed (wT[n][k] = w[k*C+n]), fp32->bf16
  auto stageW = [&](const float* w) {
#pragma unroll 4
    for (int e = 0; e < 64; ++e) {
      const int idx = tid + 256 * e;      // coalesced read
      const int k = idx >> 7, n = idx & 127;
      wT[n][k] = f2bf(w[idx]);
    }
  };

  // each wave owns N-tile = wave (16 cols), computes 4 M-tiles over K=128
  auto gemm = [&](f32x8* acc) {
    const int n0 = wave * 16;
#pragma unroll
    for (int mt = 0; mt < 4; ++mt) acc[mt] = zero8();
#pragma unroll
    for (int kk = 0; kk < C; kk += 32) {
      const bf16x16 bfrag = load_frag_b(&wT[n0][kk], C);
#pragma unroll
      for (int mt = 0; mt < 4; ++mt) {
        const bf16x16 afrag = load_frag_a(&zn[mt * 16][kk], C);
        acc[mt] = wmma_bf16(afrag, bfrag, acc[mt]);
      }
    }
  };

  const int col = wave * 16 + (lane & 15);     // output channel for this lane
  const int rhi = (lane >> 4) ? 8 : 0;         // C/D row offset for hi lanes

  // ---- one gated projection: out = mask * sigmoid(zn@Wg+bg) * (zn@Wp+bp) ----
  auto gated = [&](const float* wg, const float* bg,
                   const float* wp, const float* bp,
                   bf16_t* dst) {
    f32x8 gacc[4], pacc[4];
    stageW(wg);
    __syncthreads();
    gemm(gacc);
    __syncthreads();
    stageW(wp);
    __syncthreads();
    gemm(pacc);
    const float gb = bg[col];
    const float pb = bp[col];
#pragma unroll
    for (int mt = 0; mt < 4; ++mt) {
#pragma unroll
      for (int r = 0; r < 8; ++r) {
        const int rowl = mt * 16 + rhi + r;
        const float m  = mask[sbase + rowl];
        tmp[rowl][col] = m * sigmoidf_(gacc[mt][r] + gb) * (pacc[mt][r] + pb);
      }
    }
    __syncthreads();
    // transpose through LDS -> channel-major bf16 global store (coalesced-ish)
    {
      const int cch = tid >> 1;
      const int s0  = (tid & 1) * 32;
      bf16_t* o = dst + (size_t)cch * S + sbase + s0;
#pragma unroll 8
      for (int e = 0; e < 32; ++e) o[e] = f2bf(tmp[s0 + e][cch]);
    }
    __syncthreads();
  };

  gated(w_ag, b_ag, w_ap, b_ap, a_ws);
  gated(w_bg, b_bg, w_bp, b_bp, b_ws);

  // ---- gate g = sigmoid(zn@Wg + bg), stored [s][c] bf16 ----
  {
    f32x8 gacc[4];
    stageW(w_g);
    __syncthreads();
    gemm(gacc);
    const float gb = b_g[col];
#pragma unroll
    for (int mt = 0; mt < 4; ++mt) {
#pragma unroll
      for (int r = 0; r < 8; ++r) {
        const int rowl = mt * 16 + rhi + r;
        g_ws[(size_t)(sbase + rowl) * C + col] = f2bf(sigmoidf_(gacc[mt][r] + gb));
      }
    }
  }
}

// ---------------------------------------------------------------------------
// Kernel 2: per-channel einsum  x[c][i][j] = sum_k a[c][i][k] * b[c][j][k]
//   64x64 output tile per block; K=512 in 64-wide steps; double-buffered
//   GLOBAL_LOAD_ASYNC_TO_LDS_B128 pipeline (ASYNCcnt), one barrier per step.
// ---------------------------------------------------------------------------
__global__ __launch_bounds__(256)
void tmu_einsum_kernel(const bf16_t* __restrict__ a_ws,
                       const bf16_t* __restrict__ b_ws,
                       float* __restrict__ x_ws) {
  __shared__ __align__(16) bf16_t As[2][64][64];   // 2 x 8 KB, [M][K] slabs
  __shared__ __align__(16) bf16_t Bs[2][64][64];   // 2 x 8 KB, [N][K] slabs

  const int j0 = blockIdx.x * 64;
  const int i0 = blockIdx.y * 64;
  const int c  = blockIdx.z;
  const int tid  = threadIdx.x;
  const int wave = tid >> 5;
  const int lane = tid & 31;

  const size_t cb = (size_t)c * S;
  const unsigned long long abase =
      (unsigned long long)(const void*)(a_ws + cb);
  const unsigned long long bbase =
      (unsigned long long)(const void*)(b_ws + cb);

  const int lrow = tid >> 2;            // 0..63
  const int lcol = (tid & 3) * 16;      // 0,16,32,48 (each thread: 32B = 2xb128)

  const unsigned arow_off = (unsigned)((i0 + lrow) * N + lcol) * 2u; // bytes
  const unsigned brow_off = (unsigned)((j0 + lrow) * N + lcol) * 2u; // bytes

  // issue this thread's 2 async 2x16B copies for K-slab kk into buffer buf
  auto issue = [&](int buf, int kk) {
    const unsigned kb = (unsigned)kk * 2u;
    async_copy_32B(lds_addr_of(&As[buf][lrow][lcol]), arow_off + kb, abase);
    async_copy_32B(lds_addr_of(&Bs[buf][lrow][lcol]), brow_off + kb, bbase);
  };

  const int mt  = wave >> 1;            // M-tile 0..3
  const int ntb = (wave & 1) * 2;       // N-tile pair base: 0 or 2
  f32x8 acc0 = zero8();
  f32x8 acc1 = zero8();

  issue(0, 0);                          // prologue
  int cur = 0;
  for (int kk = 0; kk < N; kk += 64) {
    wait_async0();                      // my loads for `cur` have landed
    __syncthreads();                    // everyone's landed; prev readers done
    if (kk + 64 < N) issue(cur ^ 1, kk + 64);   // overlap with compute below
#pragma unroll
    for (int ks = 0; ks < 64; ks += 32) {
      const bf16x16 af  = load_frag_a(&As[cur][mt * 16][ks], 64);
      const bf16x16 bf0 = load_frag_b(&Bs[cur][ntb * 16][ks], 64);
      const bf16x16 bf1 = load_frag_b(&Bs[cur][(ntb + 1) * 16][ks], 64);
      acc0 = wmma_bf16(af, bf0, acc0);
      acc1 = wmma_bf16(af, bf1, acc1);
    }
    cur ^= 1;
  }

  const int rhi  = (lane >> 4) ? 8 : 0;
  const int coll = lane & 15;
#pragma unroll
  for (int r = 0; r < 8; ++r) {
    const int row = i0 + mt * 16 + rhi + r;
    x_ws[cb + (size_t)row * N + j0 + ntb * 16 + coll]       = acc0[r];
    x_ws[cb + (size_t)row * N + j0 + (ntb + 1) * 16 + coll] = acc1[r];
  }
}

// ---------------------------------------------------------------------------
// Kernel 3: LayerNorm(x) over c, out = (xn @ Wz + bz) * g
// ---------------------------------------------------------------------------
__global__ __launch_bounds__(256)
void tmu_out_kernel(const float* __restrict__ x_ws, const bf16_t* __restrict__ g_ws,
                    const float* __restrict__ ln_w, const float* __restrict__ ln_b,
                    const float* __restrict__ w_zp, const float* __restrict__ b_zp,
                    float* __restrict__ out) {
  __shared__ __align__(16) float  xb[MROWS][C];  // 32 KB gathered x (s-major)
  __shared__ __align__(16) bf16_t xn[MROWS][C];  // 16 KB LN'd bf16 (A operand)
  __shared__ __align__(16) bf16_t wT[C][C];      // 32 KB transposed Wz (B op)

  const int sbase = blockIdx.x * MROWS;
  const int tid   = threadIdx.x;
  const int wave  = tid >> 5;
  const int lane  = tid & 31;

  // gather x from channel-major layout: xb[s][c] = x_ws[c][sbase+s]
  {
    const int s  = tid & 63;
    const int cb = tid >> 6;            // 0..3
#pragma unroll 4
    for (int e = 0; e < 32; ++e) {
      const int c = cb * 32 + e;
      xb[s][c] = x_ws[(size_t)c * S + sbase + s];
    }
  }
  __syncthreads();

  // LayerNorm per spatial position (one row per wave)
#pragma unroll
  for (int it = 0; it < 8; ++it) {
    const int row = it * 8 + wave;
    float4 v = *reinterpret_cast<const float4*>(&xb[row][lane * 4]);
    float s  = v.x + v.y + v.z + v.w;
    float sq = v.x * v.x + v.y * v.y + v.z * v.z + v.w * v.w;
#pragma unroll
    for (int off = 16; off > 0; off >>= 1) {
      s  += __shfl_xor(s,  off, 32);
      sq += __shfl_xor(sq, off, 32);
    }
    const float mu  = s * (1.0f / C);
    const float var = sq * (1.0f / C) - mu * mu;
    const float rs  = rsqrtf(var + LNEPS);
    float4 w4 = *reinterpret_cast<const float4*>(ln_w + lane * 4);
    float4 b4 = *reinterpret_cast<const float4*>(ln_b + lane * 4);
    xn[row][lane * 4 + 0] = f2bf((v.x - mu) * rs * w4.x + b4.x);
    xn[row][lane * 4 + 1] = f2bf((v.y - mu) * rs * w4.y + b4.y);
    xn[row][lane * 4 + 2] = f2bf((v.z - mu) * rs * w4.z + b4.z);
    xn[row][lane * 4 + 3] = f2bf((v.w - mu) * rs * w4.w + b4.w);
  }

  // stage Wz transposed
#pragma unroll 4
  for (int e = 0; e < 64; ++e) {
    const int idx = tid + 256 * e;
    const int k = idx >> 7, n = idx & 127;
    wT[n][k] = f2bf(w_zp[idx]);
  }
  __syncthreads();

  // WMMA: 64 x 128 output, wave owns one 16-wide N-tile
  const int n0 = wave * 16;
  f32x8 acc[4];
#pragma unroll
  for (int mt = 0; mt < 4; ++mt) acc[mt] = zero8();
#pragma unroll
  for (int kk = 0; kk < C; kk += 32) {
    const bf16x16 bfrag = load_frag_b(&wT[n0][kk], C);
#pragma unroll
    for (int mt = 0; mt < 4; ++mt) {
      const bf16x16 afrag = load_frag_a(&xn[mt * 16][kk], C);
      acc[mt] = wmma_bf16(afrag, bfrag, acc[mt]);
    }
  }

  // epilogue: bias + sigmoid gate, fp32 out in [s][c] layout
  const int col = n0 + (lane & 15);
  const int rhi = (lane >> 4) ? 8 : 0;
  const float bias = b_zp[col];
#pragma unroll
  for (int mt = 0; mt < 4; ++mt) {
#pragma unroll
    for (int r = 0; r < 8; ++r) {
      const int rowl = mt * 16 + rhi + r;
      const size_t s = (size_t)(sbase + rowl);
      const float gv = bf2f(g_ws[s * C + col]);
      out[s * C + col] = (acc[mt][r] + bias) * gv;
    }
  }
}

// ---------------------------------------------------------------------------
// Launch
// ---------------------------------------------------------------------------
extern "C" void kernel_launch(void* const* d_in, const int* in_sizes, int n_in,
                              void* d_out, int out_size, void* d_ws, size_t ws_size,
                              hipStream_t stream) {
  (void)in_sizes; (void)n_in; (void)out_size; (void)ws_size;

  const float* z       = (const float*)d_in[0];
  const float* mask    = (const float*)d_in[1];
  const float* ln_in_w = (const float*)d_in[2];
  const float* ln_in_b = (const float*)d_in[3];
  const float* w_ag    = (const float*)d_in[4];
  const float* b_ag    = (const float*)d_in[5];
  const float* w_ap    = (const float*)d_in[6];
  const float* b_ap    = (const float*)d_in[7];
  const float* w_bg    = (const float*)d_in[8];
  const float* b_bg    = (const float*)d_in[9];
  const float* w_bp    = (const float*)d_in[10];
  const float* b_bp    = (const float*)d_in[11];
  const float* ln_out_w= (const float*)d_in[12];
  const float* ln_out_b= (const float*)d_in[13];
  const float* w_z     = (const float*)d_in[14];
  const float* b_z     = (const float*)d_in[15];
  const float* w_g     = (const float*)d_in[16];
  const float* b_g     = (const float*)d_in[17];

  // workspace layout (bytes): a(64MB bf16) | b(64MB bf16) | g(64MB bf16) | x(128MB f32)
  char* ws = (char*)d_ws;
  const size_t BF_BYTES = (size_t)128 * 512 * 512 * sizeof(unsigned short); // 67108864
  bf16_t* a_ws = (bf16_t*)(ws);
  bf16_t* b_ws = (bf16_t*)(ws + BF_BYTES);
  bf16_t* g_ws = (bf16_t*)(ws + 2 * BF_BYTES);
  float*  x_ws = (float*)(ws + 3 * BF_BYTES);

  const dim3 blk(256);
  tmu_proj_kernel<<<dim3(4096), blk, 0, stream>>>(
      z, mask, ln_in_w, ln_in_b, w_ag, b_ag, w_ap, b_ap,
      w_bg, b_bg, w_bp, b_bp, w_g, b_g, a_ws, b_ws, g_ws);

  tmu_einsum_kernel<<<dim3(8, 8, 128), blk, 0, stream>>>(a_ws, b_ws, x_ws);

  tmu_out_kernel<<<dim3(4096), blk, 0, stream>>>(
      x_ws, g_ws, ln_out_w, ln_out_b, w_z, b_z, (float*)d_out);
}
